// TransolverSubspaceGenerator_56916906606975
// MI455X (gfx1250) — matrix operand order repaired and checked by reading, not compiled, not run on hardware
//
#include <hip/hip_runtime.h>
#include <math.h>

constexpr int kB = 2;
constexpr int kN = 65536;
constexpr int kD = 512;
constexpr int kM = 128;
constexpr int kH = 32;

constexpr int ROWS = 128;                    // points per block
constexpr int NXS  = 520;                    // f16 elements per padded nx row
constexpr int PS   = 129;                    // float stride for p staging
constexpr int SMEM_BYTES = ROWS * NXS * 2;   // 133120 bytes (pbuf+mass alias inside)

typedef __attribute__((ext_vector_type(16))) _Float16 v16h;
typedef __attribute__((ext_vector_type(2)))  _Float16 v2h;
typedef __attribute__((ext_vector_type(8)))  float    v8f;

union V8u  { v8f v;  float f[8]; };
union F16u { int4 q[2]; v16h h; };

// A-fragment: 16-bit A layout — lanes 0..15 rows M=0..15 hold K {0..7,16..23},
// lanes 16..31 same rows hold K {8..15,24..31}. Two 16B LDS reads per lane.
__device__ __forceinline__ v16h frag_a(const _Float16* row, int k0, int c) {
  F16u f;
  f.q[0] = *(const int4*)(row + k0 + c);
  f.q[1] = *(const int4*)(row + k0 + 16 + c);
  return f.h;
}
// B-fragment: lanes 0..15 = col n, K {0..7,16..23}; lanes 16..31 = col n, K {8..15,24..31}.
__device__ __forceinline__ v16h frag_b(const _Float16* wrow, int koff) {
  F16u f;
  f.q[0] = *(const int4*)(wrow + koff);
  f.q[1] = *(const int4*)(wrow + koff + 16);
  return f.h;
}

__global__ void k_zero(float* accum) {
  if ((int)threadIdx.x < kB * kM) accum[threadIdx.x] = 0.0f;
}

__global__ void k_convert(const float* __restrict__ Wl, const float* __restrict__ Wt1,
                          _Float16* __restrict__ Wl_h, _Float16* __restrict__ Wt1_h) {
  int i = blockIdx.x * blockDim.x + threadIdx.x;
  if (i < kM * kD) Wl_h[i] = (_Float16)Wl[i];
  if (i < kH * kD) Wt1_h[i] = (_Float16)Wt1[i];
}

__global__ void k_finish(const float* __restrict__ accum, float* __restrict__ out_inv) {
  int i = threadIdx.x;
  if (i < kB * kM) out_inv[i] = 1.0f / (accum[i] + 1e-6f);
}

__global__ __launch_bounds__(256) void k_fused(
    const float* __restrict__ x, const float* __restrict__ mass,
    const float* __restrict__ ln_g, const float* __restrict__ ln_b,
    const float* __restrict__ b_lin, const float* __restrict__ W_t2,
    const float* __restrict__ b_t2,
    const _Float16* __restrict__ Wl_h, const _Float16* __restrict__ Wt1_h,
    float* __restrict__ out_tf, float* __restrict__ out_tt, float* __restrict__ accum)
{
  extern __shared__ char smem[];
  _Float16* nx     = (_Float16*)smem;                       // [128][520] f16 (phase 1-2)
  float*    pbuf   = (float*)smem;                          // [128][129] f32 (phase 3, aliased)
  float*    mass_s = (float*)(smem + (size_t)ROWS * PS * 4);

  const int tid  = threadIdx.x;
  const int lane = tid & 31;
  const int wave = tid >> 5;
  const int hlf  = lane >> 4;       // lane-half
  const int colA = lane & 15;
  const int b    = blockIdx.y;
  const int row0 = blockIdx.x * ROWS;

  // per-lane LN params for elements 2*lane + 64*i
  float g[16], bb[16];
#pragma unroll
  for (int i = 0; i < 8; ++i) {
    int e = 2 * lane + 64 * i;
    float2 tg = *(const float2*)(ln_g + e);
    float2 tb = *(const float2*)(ln_b + e);
    g[2*i] = tg.x; g[2*i+1] = tg.y;
    bb[2*i] = tb.x; bb[2*i+1] = tb.y;
  }

  // ---- Phase 1: LayerNorm -> f16 into LDS (each wave owns 16 rows) ----
  const int wr0 = row0 + wave * 16;
  for (int rr = 0; rr < 16; ++rr) {
    const float* xr = x + ((size_t)b * kN + (wr0 + rr)) * kD;
    float v[16];
    float s = 0.f, s2 = 0.f;
#pragma unroll
    for (int i = 0; i < 8; ++i) {
      float2 t = *(const float2*)(xr + 2 * lane + 64 * i);
      v[2*i] = t.x; v[2*i+1] = t.y;
      s  += t.x + t.y;
      s2 += t.x * t.x + t.y * t.y;
    }
#pragma unroll
    for (int m = 16; m >= 1; m >>= 1) {
      s  += __shfl_xor(s,  m, 32);
      s2 += __shfl_xor(s2, m, 32);
    }
    const float mean = s * (1.0f / kD);
    const float var  = s2 * (1.0f / kD) - mean * mean;
    const float rstd = rsqrtf(var + 1e-5f);
    _Float16* nrow = nx + (size_t)(wave * 16 + rr) * NXS;
#pragma unroll
    for (int i = 0; i < 8; ++i) {
      float a0 = (v[2*i]   - mean) * rstd * g[2*i]   + bb[2*i];
      float a1 = (v[2*i+1] - mean) * rstd * g[2*i+1] + bb[2*i+1];
      v2h hv; hv.x = (_Float16)a0; hv.y = (_Float16)a1;
      *(v2h*)(nrow + 2 * lane + 64 * i) = hv;
    }
  }

  // ---- Phase 2: WMMA — logits (8 tiles of 16 modes) + temp hidden (2 tiles) ----
  V8u acc[8], ht[2];
#pragma unroll
  for (int t = 0; t < 8; ++t)
#pragma unroll
    for (int e = 0; e < 8; ++e) acc[t].f[e] = 0.f;
#pragma unroll
  for (int t = 0; t < 2; ++t)
#pragma unroll
    for (int e = 0; e < 8; ++e) ht[t].f[e] = 0.f;

  const _Float16* arow = nx + (size_t)(wave * 16 + colA) * NXS;
  const int ca = hlf * 8;    // A K-offset within chunk
  const int kb = hlf * 8;    // B K-offset within chunk

  for (int k0 = 0; k0 < kD; k0 += 32) {
    v16h a = frag_a(arow, k0, ca);
#pragma unroll
    for (int t = 0; t < 2; ++t) {
      v16h bm = frag_b(Wt1_h + (size_t)(t * 16 + colA) * kD, k0 + kb);
      ht[t].v = __builtin_amdgcn_wmma_f32_16x16x32_f16(false, a, false, bm, (short)0, ht[t].v, false, false);
    }
#pragma unroll
    for (int t = 0; t < 8; ++t) {
      v16h bm = frag_b(Wl_h + (size_t)(t * 16 + colA) * kD, k0 + kb);
      acc[t].v = __builtin_amdgcn_wmma_f32_16x16x32_f16(false, a, false, bm, (short)0, acc[t].v, false, false);
    }
  }

  // ---- temperature: GELU(exact) -> dot(W_t2) -> softplus -> clamp ----
  const float wt2a = W_t2[colA];
  const float wt2b = W_t2[16 + colA];
  const float bt2  = b_t2[0];
  float invtau[8];
#pragma unroll
  for (int vv = 0; vv < 8; ++vv) {
    float h0 = ht[0].f[vv], h1 = ht[1].f[vv];
    float g0 = 0.5f * h0 * (1.0f + erff(h0 * 0.70710678118f));
    float g1 = 0.5f * h1 * (1.0f + erff(h1 * 0.70710678118f));
    float part = wt2a * g0 + wt2b * g1;
#pragma unroll
    for (int m = 8; m >= 1; m >>= 1) part += __shfl_xor(part, m, 32);
    float z  = part + bt2;
    float sp = (z > 20.0f) ? z : log1pf(expf(z));
    float tau = fminf(fmaxf(sp, 0.01f), 3.0f);
    invtau[vv] = 1.0f / tau;
  }

  // ---- softmax((logits+b)/tau) over M, store trial_func from registers ----
  float bl[8];
#pragma unroll
  for (int t = 0; t < 8; ++t) bl[t] = b_lin[t * 16 + colA];

#pragma unroll
  for (int vv = 0; vv < 8; ++vv) {
    float sv[8];
    float mx = -INFINITY;
#pragma unroll
    for (int t = 0; t < 8; ++t) {
      sv[t] = (acc[t].f[vv] + bl[t]) * invtau[vv];
      mx = fmaxf(mx, sv[t]);
    }
#pragma unroll
    for (int m = 8; m >= 1; m >>= 1) mx = fmaxf(mx, __shfl_xor(mx, m, 32));
    float sum = 0.f;
#pragma unroll
    for (int t = 0; t < 8; ++t) { sv[t] = expf(sv[t] - mx); sum += sv[t]; }
#pragma unroll
    for (int m = 8; m >= 1; m >>= 1) sum += __shfl_xor(sum, m, 32);
    const float inv = 1.0f / sum;
    const int grow = wr0 + vv + 8 * hlf;
    float* o = out_tf + ((size_t)b * kN + grow) * kM;
#pragma unroll
    for (int t = 0; t < 8; ++t) {
      float p = sv[t] * inv;
      acc[t].f[vv] = p;               // keep for transpose stage
      o[t * 16 + colA] = p;           // 16 lanes -> one 64B line
    }
  }

  __syncthreads();                    // all nx reads done; alias LDS as pbuf
  if (tid < ROWS) mass_s[tid] = mass[(size_t)b * kN + row0 + tid];
#pragma unroll
  for (int vv = 0; vv < 8; ++vv) {
    const int rl = wave * 16 + vv + 8 * hlf;
#pragma unroll
    for (int t = 0; t < 8; ++t) pbuf[rl * PS + t * 16 + colA] = acc[t].f[vv];
  }
  __syncthreads();

  // ---- trial_in_t (B,M,N): coalesced 512B runs per mode ----
  {
    const int r  = tid & 127;
    const int mo = tid >> 7;          // 0 or 1
    const float ms = mass_s[r];
    for (int c0 = 0; c0 < kM; c0 += 2) {
      const int m = c0 + mo;
      out_tt[((size_t)b * kM + m) * kN + row0 + r] = pbuf[r * PS + m] * ms;
    }
  }

  // ---- per-mode partial sums over this block's rows ----
  if (tid < kM) {
    float s = 0.f;
    for (int r = 0; r < ROWS; ++r) s += pbuf[r * PS + tid] * mass_s[r];
    atomicAdd(&accum[b * kM + tid], s);
  }
}

extern "C" void kernel_launch(void* const* d_in, const int* in_sizes, int n_in,
                              void* d_out, int out_size, void* d_ws, size_t ws_size,
                              hipStream_t stream) {
  (void)in_sizes; (void)n_in; (void)out_size; (void)ws_size;
  const float* x     = (const float*)d_in[0];
  const float* mass  = (const float*)d_in[1];
  const float* ln_g  = (const float*)d_in[2];
  const float* ln_b  = (const float*)d_in[3];
  const float* W_lin = (const float*)d_in[4];
  const float* b_lin = (const float*)d_in[5];
  const float* W_t1  = (const float*)d_in[6];
  const float* W_t2  = (const float*)d_in[7];
  const float* b_t2  = (const float*)d_in[8];

  _Float16* Wl_h  = (_Float16*)d_ws;                          // 128*512*2 = 131072 B
  _Float16* Wt1_h = (_Float16*)((char*)d_ws + 131072);        //  32*512*2 =  32768 B
  float*    accum = (float*)((char*)d_ws + 163840);           //  256*4    =   1024 B

  float* out     = (float*)d_out;
  float* out_tf  = out;                                       // (B,N,M)
  float* out_tt  = out + (size_t)kB * kN * kM;                // (B,M,N)
  float* out_inv = out + 2 * (size_t)kB * kN * kM;            // (B,M,1)

  hipFuncSetAttribute(reinterpret_cast<const void*>(k_fused),
                      hipFuncAttributeMaxDynamicSharedMemorySize, SMEM_BYTES);

  hipLaunchKernelGGL(k_zero, dim3(1), dim3(256), 0, stream, accum);
  hipLaunchKernelGGL(k_convert, dim3((kM * kD + 255) / 256), dim3(256), 0, stream,
                     W_lin, W_t1, Wl_h, Wt1_h);
  hipLaunchKernelGGL(k_fused, dim3(kN / ROWS, kB), dim3(256), SMEM_BYTES, stream,
                     x, mass, ln_g, ln_b, b_lin, W_t2, b_t2, Wl_h, Wt1_h,
                     out_tf, out_tt, accum);
  hipLaunchKernelGGL(k_finish, dim3(1), dim3(256), 0, stream, accum, out_inv);
}
